// Segger_67894843015286
// MI455X (gfx1250) — compile-verified, matrix operand order checked by reading, not run on hardware
//
#include <hip/hip_runtime.h>
#include <hip/hip_bf16.h>
#include <math.h>

// ---------------------------------------------------------------------------
// GATv2 x2 for MI455X (gfx1250, wave32).
//  - Dense transforms: v_wmma_f32_16x16x32_f16, fully unrolled (K and N tile
//    counts are template constants). One wave owns a 16-row strip and all
//    N-tiles; A fragments via contiguous float4 loads, B pre-packed to f16
//    fragment layout so the hot loop is pure load/cvt/wmma straight-line code.
//  - Edge phase: wave-per-edge coalesced gather + shfl reduction, segment
//    softmax via float atomicMax/atomicAdd, wave-per-edge coalesced scatter.
// ---------------------------------------------------------------------------

typedef __attribute__((ext_vector_type(16))) _Float16 v16h;
typedef __attribute__((ext_vector_type(8)))  float    v8f;

#define HEADS 3
#define CH    32          // HID == OUT == 32  -> one lane per channel
#define HC    96          // HEADS * CH
#define NEG_SLOPE 0.2f
#define EPS_F 1e-16f

// ---------------------------------------------------------------------------
// Pack B[K x Ncols] (f32, row-major) into f16 WMMA fragments:
//   Bpack[((kstep*ntilesN + tileN)*32 + lane) * 16 + i]
// where element i=2v+h of lane (ncol=tileN*16+(lane&15), khalf=lane>>4) is
//   k = kstep*32 + (v>>2)*16 + khalf*8 + (v&3)*2 + h   (0 if k >= K).
// Tiny kernel (weights are <= 96x96); runs once per GEMM.
// ---------------------------------------------------------------------------
__global__ __launch_bounds__(32)
void gatv2_pack_b(const float* __restrict__ B, _Float16* __restrict__ Bpack,
                  int K, int Ncols)
{
    const int kstep  = blockIdx.x;
    const int tileN  = blockIdx.y;
    const int ntiles = gridDim.y;
    const int lane   = threadIdx.x;
    const int ncol   = tileN * 16 + (lane & 15);
    const int khalf  = lane >> 4;

    v16h frag;
#pragma unroll
    for (int i = 0; i < 16; ++i) {
        const int v = i >> 1, h = i & 1;
        const int k = kstep * 32 + ((v >> 2) << 4) + (khalf << 3) + ((v & 3) << 1) + h;
        frag[i] = (_Float16)((k < K) ? B[(size_t)k * Ncols + ncol] : 0.f);
    }
    const size_t entry = ((size_t)(kstep * ntiles + tileN) * 32 + lane);
    *reinterpret_cast<v16h*>(Bpack + entry * 16) = frag;
}

// ---------------------------------------------------------------------------
// C[M x Ncols] = A[M x KVAL] @ B + bias1 + bias2 (optional), optional ReLU.
// NTILES = Ncols/16, KVAL = K (both compile-time -> fully unrolled, no
// per-iteration control flow). One wave per 16-row strip, 4 waves/block.
// ---------------------------------------------------------------------------
template <int NTILES, int KVAL>
__global__ __launch_bounds__(128)
void gatv2_gemm_wmma(const float* __restrict__ A, const _Float16* __restrict__ Bpack,
                     const float* __restrict__ bias1, const float* __restrict__ bias2,
                     float* __restrict__ C, int M, int do_relu)
{
    const int wave   = threadIdx.x >> 5;
    const int lane   = threadIdx.x & 31;
    const int tileM  = blockIdx.x * 4 + wave;
    if (tileM * 16 >= M) return;

    constexpr int Ncols = NTILES * 16;
    const int l15   = lane & 15;
    const int khalf = lane >> 4;
    const int mrow  = tileM * 16 + l15;

    const float4* __restrict__ Arow =
        reinterpret_cast<const float4*>(A + (size_t)mrow * KVAL);
    const v16h* __restrict__ Bf = reinterpret_cast<const v16h*>(Bpack);

    v8f acc[NTILES];
#pragma unroll
    for (int t = 0; t < NTILES; ++t) acc[t] = (v8f){};

#pragma unroll
    for (int k0 = 0; k0 < KVAL; k0 += 32) {
        // ---- A fragment: two contiguous 8-float runs per lane -------------
        const int base0 = (k0 + (khalf << 3)) >> 2;     // float4 units
        v16h a;
        {
            const float4 r0 = Arow[base0];
            const float4 r1 = Arow[base0 + 1];
            const float rr[8] = {r0.x, r0.y, r0.z, r0.w, r1.x, r1.y, r1.z, r1.w};
#pragma unroll
            for (int i = 0; i < 8; ++i) a[i] = (_Float16)rr[i];
        }
        if constexpr (true) {
            if ((k0 + 16) < KVAL) {                      // folds at compile time
                const float4 r0 = Arow[base0 + 4];
                const float4 r1 = Arow[base0 + 5];
                const float rr[8] = {r0.x, r0.y, r0.z, r0.w, r1.x, r1.y, r1.z, r1.w};
#pragma unroll
                for (int i = 0; i < 8; ++i) a[8 + i] = (_Float16)rr[i];
            } else {
#pragma unroll
                for (int i = 0; i < 8; ++i) a[8 + i] = (_Float16)0.f;
            }
        }
        // ---- all N tiles reuse this A fragment ---------------------------
        const size_t kbase = (size_t)(k0 >> 5) * NTILES;
#pragma unroll
        for (int t = 0; t < NTILES; ++t) {
            const v16h b = Bf[(kbase + t) * 32 + lane];
            acc[t] = __builtin_amdgcn_wmma_f32_16x16x32_f16(
                false, a, false, b, (short)0, acc[t], false, false);
        }
    }

    // ---- epilogue: bias + optional ReLU, coalesced row stores -------------
#pragma unroll
    for (int t = 0; t < NTILES; ++t) {
        const int n = t * 16 + l15;
        float bsum = 0.f;
        if (bias1) bsum += bias1[n];
        if (bias2) bsum += bias2[n];
#pragma unroll
        for (int r = 0; r < 8; ++r) {
            const int m = tileM * 16 + r + (khalf << 3);
            float v = acc[t][r] + bsum;
            if (do_relu) v = v > 0.f ? v : 0.f;
            C[(size_t)m * Ncols + n] = v;
        }
    }
}

// ---------------------------------------------------------------------------
// Segment-softmax scratch init: max = -inf, denom = 0.
// ---------------------------------------------------------------------------
__global__ void gatv2_init_seg(float* __restrict__ maxb, float* __restrict__ denom, int n)
{
    int i = blockIdx.x * blockDim.x + threadIdx.x;
    if (i < n) { maxb[i] = -3.0e38f; denom[i] = 0.f; }
}

// ---------------------------------------------------------------------------
// Pass 1 (wave per edge, lane == channel): coalesced gather + shfl reduce.
// score[e,h] = sum_c att[h,c] * leaky_relu(hl[src,h,c] + hr[dst,h,c])
// ---------------------------------------------------------------------------
__global__ void gatv2_edge_score(const int* __restrict__ src_idx,
                                 const int* __restrict__ dst_idx,
                                 const float* __restrict__ hl,
                                 const float* __restrict__ hr,
                                 const float* __restrict__ att,   // [HEADS*CH]
                                 float* __restrict__ score,       // [E*HEADS]
                                 float* __restrict__ maxb,        // [N*HEADS]
                                 int nedges)
{
    const int e    = (int)((blockIdx.x * (size_t)blockDim.x + threadIdx.x) >> 5);
    const int lane = threadIdx.x & 31;
    if (e >= nedges) return;
    const int s = src_idx[e];
    const int d = dst_idx[e];
    const float* __restrict__ pl = hl + (size_t)s * HC;
    const float* __restrict__ pr = hr + (size_t)d * HC;
    __builtin_prefetch(pl, 0, 0);    // global_prefetch_b8
    __builtin_prefetch(pr, 0, 0);
#pragma unroll
    for (int h = 0; h < HEADS; ++h) {
        float v = pl[h * CH + lane] + pr[h * CH + lane];
        v = v > 0.f ? v : v * NEG_SLOPE;
        v *= att[h * CH + lane];
        // wave32 all-lanes sum
#pragma unroll
        for (int off = 16; off > 0; off >>= 1)
            v += __shfl_xor(v, off, 32);
        if (lane == 0) {
            score[(size_t)e * HEADS + h] = v;
            atomicMax(&maxb[(size_t)d * HEADS + h], v);  // global_atomic_max_num_f32
        }
    }
}

// ---------------------------------------------------------------------------
// Pass 2: p = exp(score - max[dst]); denom[dst] += p (in-place over score).
// ---------------------------------------------------------------------------
__global__ void gatv2_edge_exp(const int* __restrict__ dst_idx,
                               float* __restrict__ score,
                               const float* __restrict__ maxb,
                               float* __restrict__ denom,
                               int nedges)
{
    int e = blockIdx.x * blockDim.x + threadIdx.x;
    if (e >= nedges) return;
    const int d = dst_idx[e];
#pragma unroll
    for (int h = 0; h < HEADS; ++h) {
        float p = __expf(score[(size_t)e * HEADS + h] - maxb[(size_t)d * HEADS + h]);
        score[(size_t)e * HEADS + h] = p;
        atomicAdd(&denom[(size_t)d * HEADS + h], p);
    }
}

// ---------------------------------------------------------------------------
// Pass 3 (wave per edge, lane == channel): coalesced scatter-add.
// out[dst, h, lane] += hl[src, h, lane] * p/(denom+eps)
// ---------------------------------------------------------------------------
__global__ void gatv2_edge_scatter(const int* __restrict__ src_idx,
                                   const int* __restrict__ dst_idx,
                                   const float* __restrict__ hl,
                                   const float* __restrict__ score,
                                   const float* __restrict__ denom,
                                   float* __restrict__ out,
                                   int nedges)
{
    const int e    = (int)((blockIdx.x * (size_t)blockDim.x + threadIdx.x) >> 5);
    const int lane = threadIdx.x & 31;
    if (e >= nedges) return;
    const int s = src_idx[e];
    const int d = dst_idx[e];
    const float* __restrict__ pl = hl + (size_t)s * HC;
    __builtin_prefetch(pl, 0, 0);
#pragma unroll
    for (int h = 0; h < HEADS; ++h) {
        const float alpha = score[(size_t)e * HEADS + h]
                          / (denom[(size_t)d * HEADS + h] + EPS_F);
        const float m = pl[h * CH + lane] * alpha;
        atomicAdd(&out[(size_t)d * HC + h * CH + lane], m);
    }
}

// ---------------------------------------------------------------------------
// Elementwise ReLU (between layers).
// ---------------------------------------------------------------------------
__global__ void gatv2_relu(float* __restrict__ x, size_t n)
{
    size_t i = blockIdx.x * (size_t)blockDim.x + threadIdx.x;
    if (i < n) { float v = x[i]; x[i] = v > 0.f ? v : 0.f; }
}

// ---------------------------------------------------------------------------
// Host-side orchestration (all launches on `stream`, graph-capture safe).
// pack_b -> gemm pairs are stream-ordered, so one Bpack buffer is reused.
// ---------------------------------------------------------------------------
extern "C" void kernel_launch(void* const* d_in, const int* in_sizes, int n_in,
                              void* d_out, int out_size, void* d_ws, size_t ws_size,
                              hipStream_t stream)
{
    // setup_inputs() order
    const float* x     = (const float*)d_in[0];
    const int*   ei    = (const int*)  d_in[1];   // [2,E] row-major: src then dst
    const float* W0    = (const float*)d_in[2];
    const float* Wl1   = (const float*)d_in[3];
    const float* bl1   = (const float*)d_in[4];
    const float* Wr1   = (const float*)d_in[5];
    const float* br1   = (const float*)d_in[6];
    const float* att1  = (const float*)d_in[7];
    const float* b1    = (const float*)d_in[8];
    const float* Wf    = (const float*)d_in[9];
    const float* bf    = (const float*)d_in[10];
    const float* Wl2   = (const float*)d_in[11];
    const float* bl2   = (const float*)d_in[12];
    const float* Wr2   = (const float*)d_in[13];
    const float* br2   = (const float*)d_in[14];
    const float* att2  = (const float*)d_in[15];
    const float* b2    = (const float*)d_in[16];
    const float* Wlast = (const float*)d_in[17];
    const float* blast = (const float*)d_in[18];
    float*       out   = (float*)d_out;

    const int F_IN = 128, INIT_EMB = 16;
    const int Nn = in_sizes[0] / F_IN;        // 100000 (multiple of 16)
    const int Ee = in_sizes[1] / 2;           // 1600000
    const int* src = ei;
    const int* dst = ei + Ee;

    // workspace layout (floats, then 32B-aligned f16 pack buffer)
    float* ws    = (float*)d_ws;
    float* h0    = ws;                              // N * 16
    float* h1    = h0   + (size_t)Nn * INIT_EMB;    // N * 96
    float* hl    = h1   + (size_t)Nn * HC;          // N * 96
    float* hr    = hl   + (size_t)Nn * HC;          // N * 96
    float* score = hr   + (size_t)Nn * HC;          // E * 3
    float* maxb  = score + (size_t)Ee * HEADS;      // N * 3
    float* denom = maxb + (size_t)Nn * HEADS;       // N * 3
    uintptr_t bp = (uintptr_t)(denom + (size_t)Nn * HEADS);
    bp = (bp + 31u) & ~(uintptr_t)31u;
    _Float16* Bpack = (_Float16*)bp;                // <= 9216 halves used

    const dim3 wblk(32);
    const dim3 gblk(128);                           // 4 waves / block
    const int  mtiles  = Nn / 16;
    const int  gblocks = (mtiles + 3) / 4;
    const int  TPB = 256;
    const int  eblocks = (Ee + TPB - 1) / TPB;
    const int  wavblocks = (int)(((size_t)Ee * 32 + TPB - 1) / TPB);
    const int  segN = Nn * HEADS;

    // ---- lin0: h0 = relu(x @ W0)  (K=128, N=16) --------------------------
    gatv2_pack_b<<<dim3(F_IN / 32, 1), wblk, 0, stream>>>(W0, Bpack, F_IN, INIT_EMB);
    gatv2_gemm_wmma<1, 128><<<gblocks, gblk, 0, stream>>>(
        x, Bpack, nullptr, nullptr, h0, Nn, 1);

    // ---- layer 1 transforms (K=16 -> 1 kstep, N=96 -> 6 tiles) -----------
    gatv2_pack_b<<<dim3(1, HC / 16), wblk, 0, stream>>>(Wl1, Bpack, INIT_EMB, HC);
    gatv2_gemm_wmma<6, 16><<<gblocks, gblk, 0, stream>>>(
        h0, Bpack, bl1, nullptr, hl, Nn, 0);
    gatv2_pack_b<<<dim3(1, HC / 16), wblk, 0, stream>>>(Wr1, Bpack, INIT_EMB, HC);
    gatv2_gemm_wmma<6, 16><<<gblocks, gblk, 0, stream>>>(
        h0, Bpack, br1, nullptr, hr, Nn, 0);
    // skip connection seeds the aggregation buffer: h1 = h0@Wf + bf + b1
    gatv2_pack_b<<<dim3(1, HC / 16), wblk, 0, stream>>>(Wf, Bpack, INIT_EMB, HC);
    gatv2_gemm_wmma<6, 16><<<gblocks, gblk, 0, stream>>>(
        h0, Bpack, bf, b1, h1, Nn, 0);

    // ---- layer 1 edge phase ---------------------------------------------
    gatv2_init_seg<<<(segN + TPB - 1) / TPB, TPB, 0, stream>>>(maxb, denom, segN);
    gatv2_edge_score<<<wavblocks, TPB, 0, stream>>>(src, dst, hl, hr, att1, score, maxb, Ee);
    gatv2_edge_exp<<<eblocks, TPB, 0, stream>>>(dst, score, maxb, denom, Ee);
    gatv2_edge_scatter<<<wavblocks, TPB, 0, stream>>>(src, dst, hl, score, denom, h1, Ee);
    gatv2_relu<<<(int)(((size_t)Nn * HC + TPB - 1) / TPB), TPB, 0, stream>>>(h1, (size_t)Nn * HC);

    // ---- layer 2 transforms (K=96 -> 3 ksteps, N=96 -> 6 tiles) ----------
    gatv2_pack_b<<<dim3(3, HC / 16), wblk, 0, stream>>>(Wl2, Bpack, HC, HC);
    gatv2_gemm_wmma<6, 96><<<gblocks, gblk, 0, stream>>>(
        h1, Bpack, bl2, nullptr, hl, Nn, 0);
    gatv2_pack_b<<<dim3(3, HC / 16), wblk, 0, stream>>>(Wr2, Bpack, HC, HC);
    gatv2_gemm_wmma<6, 96><<<gblocks, gblk, 0, stream>>>(
        h1, Bpack, br2, nullptr, hr, Nn, 0);
    // seed output with skip: out = h1@Wlast + blast + b2
    gatv2_pack_b<<<dim3(3, HC / 16), wblk, 0, stream>>>(Wlast, Bpack, HC, HC);
    gatv2_gemm_wmma<6, 96><<<gblocks, gblk, 0, stream>>>(
        h1, Bpack, blast, b2, out, Nn, 0);

    // ---- layer 2 edge phase (no final relu) ------------------------------
    gatv2_init_seg<<<(segN + TPB - 1) / TPB, TPB, 0, stream>>>(maxb, denom, segN);
    gatv2_edge_score<<<wavblocks, TPB, 0, stream>>>(src, dst, hl, hr, att2, score, maxb, Ee);
    gatv2_edge_exp<<<eblocks, TPB, 0, stream>>>(dst, score, maxb, denom, Ee);
    gatv2_edge_scatter<<<wavblocks, TPB, 0, stream>>>(src, dst, hl, score, denom, out, Ee);
}
